// Graph_Representation_24077586661648
// MI455X (gfx1250) — matrix operand order repaired and checked by reading, not compiled, not run on hardware
//
#include <hip/hip_runtime.h>
#include <hip/hip_bf16.h>

// ---------------------------------------------------------------------------
// Problem constants (from reference)
// ---------------------------------------------------------------------------
#define NNODES 50000
#define NEDGES 800000
#define DD     64          // node feature dim D
#define EDD    32          // edge feature dim
#define GG     50          // output dim
#define K1     160         // 2*D + ED

#define FE_BLOB_ELEMS   (DD * K1 + DD * DD)   // 14336 bf16 (W1^T | W2^T)
#define LSTM_BLOB_ELEMS (256 * 128)           // 32768 bf16 (stacked [Wih;Whh]^T)

typedef __attribute__((ext_vector_type(16))) __bf16 v16bf;
typedef __attribute__((ext_vector_type(8)))  __bf16 v8bf;
typedef __attribute__((ext_vector_type(8)))  float  v8f;

__device__ __forceinline__ __bf16 to_bf16(float f) { return (__bf16)f; }

__device__ __forceinline__ v8f wmma_bf16(v16bf a, v16bf b, v8f c) {
  // D = A(16x32 bf16) * B(32x16 bf16) + C(16x16 f32)
  return __builtin_amdgcn_wmma_f32_16x16x32_bf16(
      /*neg_a=*/false, a, /*neg_b=*/false, b,
      /*c_mod=*/(short)0, c, /*reuse_a=*/false, /*reuse_b=*/false);
}

__device__ __forceinline__ float fast_sigmoid(float v) {
  return 1.0f / (1.0f + __expf(-v));               // overflow-safe
}
__device__ __forceinline__ float fast_tanh(float v) {
  return 1.0f - 2.0f / (__expf(2.0f * v) + 1.0f);  // safe at both infinities
}

// ---- CDNA5 async global->LDS copy (ASYNCcnt path, 16B per lane) -----------
__device__ __forceinline__ void async_copy_b128(void* lds_dst, const void* g_src) {
  unsigned l = (unsigned)(uintptr_t)lds_dst;                 // LDS aperture: addr[31:0]
  unsigned long long g = (unsigned long long)(uintptr_t)g_src;
  asm volatile("global_load_async_to_lds_b128 %0, %1, off"
               :: "v"(l), "v"(g) : "memory");
}
__device__ __forceinline__ void wait_async0() {
  asm volatile("s_wait_asynccnt 0" ::: "memory");
}

// ---------------------------------------------------------------------------
// Init helpers
// ---------------------------------------------------------------------------
__global__ void init_state_kernel(const float* __restrict__ xin,
                                  float* __restrict__ xcur,
                                  float* __restrict__ cbuf, int n) {
  int i = blockIdx.x * blockDim.x + threadIdx.x;
  if (i < n) { xcur[i] = xin[i]; cbuf[i] = 0.0f; }
}

__global__ void zero_kernel(float* __restrict__ p, int n) {
  int i = blockIdx.x * blockDim.x + threadIdx.x;
  if (i < n) p[i] = 0.0f;
}

// One-shot f32 -> transposed-bf16 weight conversion for one MP step.
__global__ void convert_weights_kernel(const float* __restrict__ W1,
                                       const float* __restrict__ W2,
                                       const float* __restrict__ Wih,
                                       const float* __restrict__ Whh,
                                       __bf16* __restrict__ feblob,
                                       __bf16* __restrict__ lstmblob) {
  int i = blockIdx.x * blockDim.x + threadIdx.x;
  if (i < DD * K1) {                       // W1^T: [n][k], n in 0..63, k in 0..159
    int n = i / K1, k = i - n * K1;
    feblob[i] = to_bf16(W1[k * DD + n]);
  } else if (i < FE_BLOB_ELEMS) {          // W2^T: [n][k], 64x64
    int j = i - DD * K1;
    int n = j >> 6, k = j & 63;
    feblob[i] = to_bf16(W2[k * DD + n]);
  }
  if (i < LSTM_BLOB_ELEMS) {               // stacked [Wih;Whh]^T: [n][k], 256x128
    int n = i >> 7, k = i & 127;
    float v = (k < DD) ? Wih[k * 256 + n] : Whh[(k - DD) * 256 + n];
    lstmblob[i] = to_bf16(v);
  }
}

// ---------------------------------------------------------------------------
// Edge MLP + scatter-add (segment_sum):  m = relu([x_d|x_s|ea] @ W1 + b1) @ W2 + b2
// One wave == 16 edges.  bf16 WMMA, f32 accumulate.  Weights arrive via
// global_load_async_to_lds_b128 (pre-converted bf16, L2-resident).
// ---------------------------------------------------------------------------
__launch_bounds__(128)
__global__ void edge_mpn_kernel(const float* __restrict__ x,
                                const float* __restrict__ edge_attr,
                                const int*   __restrict__ src_idx,
                                const int*   __restrict__ dst_idx,
                                const __bf16* __restrict__ wblob,   // W1^T | W2^T bf16
                                const float* __restrict__ b1,
                                const float* __restrict__ b2,
                                float* __restrict__ agg) {
  __shared__ __attribute__((aligned(32))) __bf16 sWblob[FE_BLOB_ELEMS];
  __shared__ float sB1[DD];
  __shared__ float sB2[DD];
  __shared__ __attribute__((aligned(32))) __bf16 sY[4][16 * DD];  // per-wave Y1 staging

  const int tid = threadIdx.x;

  // Async DMA the 28KB bf16 weight blob into LDS (no VGPR round-trip).
  {
    char* lbase = (char*)sWblob;
    const char* gbase = (const char*)wblob;
#pragma unroll
    for (int ofs = 0; ofs < FE_BLOB_ELEMS * 2; ofs += 128 * 16) {
      async_copy_b128(lbase + ofs + tid * 16, gbase + ofs + tid * 16);
    }
  }
  if (tid < DD) { sB1[tid] = b1[tid]; sB2[tid] = b2[tid]; }
  wait_async0();          // this wave's async stores to LDS are complete
  __syncthreads();        // all waves' portions visible

  const __bf16* sW1t = sWblob;            // [n][k] row stride K1
  const __bf16* sW2t = sWblob + DD * K1;  // [n][k] row stride DD

  const int wave  = tid >> 5;
  const int lane  = tid & 31;
  const int half  = lane >> 4;    // K-half for A/B fragments
  const int m     = lane & 15;    // row (A) / column (B,C)
  const int ebase = (blockIdx.x * 4 + wave) * 16;
  const int e     = ebase + m;

  const int s = src_idx[e];
  const int d = dst_idx[e];
  const float* rowD = x + (size_t)d * DD;
  const float* rowS = x + (size_t)s * DD;
  const float* rowE = edge_attr + (size_t)e * EDD;

  v8f zf;
#pragma unroll
  for (int j = 0; j < 8; ++j) zf[j] = 0.0f;

  // ---------------- GEMM1: [16 x 160] x [160 x 64] ----------------
  v8f acc[4];
#pragma unroll
  for (int nt = 0; nt < 4; ++nt) acc[nt] = zf;

#pragma unroll
  for (int kc = 0; kc < 5; ++kc) {
    const float* base = (kc < 2) ? rowD : ((kc < 4) ? rowS : rowE);
    const int kb = (kc < 2) ? kc * 32 : ((kc < 4) ? (kc - 2) * 32 : 0);

    // A fragment: lane holds row m, K = {kb+half*16 .. +15}
    const float4 q0 = *(const float4*)(base + kb + half * 16);
    const float4 q1 = *(const float4*)(base + kb + half * 16 + 4);
    const float4 r0 = *(const float4*)(base + kb + half * 16 + 8);
    const float4 r1 = *(const float4*)(base + kb + half * 16 + 12);
    v16bf a;
    a[0]  = to_bf16(q0.x); a[1]  = to_bf16(q0.y); a[2]  = to_bf16(q0.z); a[3]  = to_bf16(q0.w);
    a[4]  = to_bf16(q1.x); a[5]  = to_bf16(q1.y); a[6]  = to_bf16(q1.z); a[7]  = to_bf16(q1.w);
    a[8]  = to_bf16(r0.x); a[9]  = to_bf16(r0.y); a[10] = to_bf16(r0.z); a[11] = to_bf16(r0.w);
    a[12] = to_bf16(r1.x); a[13] = to_bf16(r1.y); a[14] = to_bf16(r1.z); a[15] = to_bf16(r1.w);

#pragma unroll
    for (int nt = 0; nt < 4; ++nt) {
      const v16bf b = *(const v16bf*)&sW1t[(nt * 16 + m) * K1 + kc * 32 + half * 16];
      acc[nt] = wmma_bf16(a, b, acc[nt]);
    }
  }

  // bias + ReLU, transpose via per-wave LDS tile (C layout: lane=col, rows j+8*half)
  __bf16* yw = sY[wave];
#pragma unroll
  for (int nt = 0; nt < 4; ++nt) {
    const float bias = sB1[nt * 16 + m];
#pragma unroll
    for (int j = 0; j < 8; ++j) {
      float v = acc[nt][j] + bias;
      v = v > 0.0f ? v : 0.0f;
      yw[(j + 8 * half) * DD + nt * 16 + m] = to_bf16(v);
    }
  }
  asm volatile("s_wait_dscnt 0" ::: "memory");   // wave-local LDS RAW fence

  // ---------------- GEMM2: [16 x 64] x [64 x 64] ----------------
  v8f acc2[4];
#pragma unroll
  for (int nt = 0; nt < 4; ++nt) acc2[nt] = zf;

#pragma unroll
  for (int kc = 0; kc < 2; ++kc) {
    const v8bf lo = *(const v8bf*)&yw[m * DD + kc * 32 + half * 16];
    const v8bf hi = *(const v8bf*)&yw[m * DD + kc * 32 + half * 16 + 8];
    const v16bf a = __builtin_shufflevector(lo, hi, 0, 1, 2, 3, 4, 5, 6, 7,
                                            8, 9, 10, 11, 12, 13, 14, 15);
#pragma unroll
    for (int nt = 0; nt < 4; ++nt) {
      const v16bf b = *(const v16bf*)&sW2t[(nt * 16 + m) * DD + kc * 32 + half * 16];
      acc2[nt] = wmma_bf16(a, b, acc2[nt]);
    }
  }

  // Scatter segment-sum: lane holds column (nt*16+m), rows j+8*half
  int drow[8];
#pragma unroll
  for (int j = 0; j < 8; ++j) drow[j] = dst_idx[ebase + j + 8 * half];

#pragma unroll
  for (int nt = 0; nt < 4; ++nt) {
    const float bias2 = sB2[nt * 16 + m];
#pragma unroll
    for (int j = 0; j < 8; ++j) {
      atomicAdd(&agg[(size_t)drow[j] * DD + nt * 16 + m], acc2[nt][j] + bias2);
    }
  }
}

// ---------------------------------------------------------------------------
// LSTM cell: gates = x@Wih + a@Whh + (bih+bhh); fused as [x|a](16x128) @ Wstk(128x256)
// One wave == 16 nodes.  Gate tiles grouped (t, t+4, t+8, t+12) so i/f/g/o for the
// same (node, feature) land in the same lane/VGPR slot.
// ---------------------------------------------------------------------------
__launch_bounds__(128)
__global__ void lstm_kernel(float* __restrict__ x,          // in: x, out: h
                            const float* __restrict__ a,    // aggregated messages
                            float* __restrict__ c,
                            const __bf16* __restrict__ wblob,  // [Wih;Whh]^T bf16
                            const float* __restrict__ bih,
                            const float* __restrict__ bhh,
                            int ntiles) {
  __shared__ __attribute__((aligned(32))) __bf16 sWt[LSTM_BLOB_ELEMS];  // [n][k]
  __shared__ float sB[256];

  const int tid = threadIdx.x;
  {
    char* lbase = (char*)sWt;
    const char* gbase = (const char*)wblob;
#pragma unroll
    for (int ofs = 0; ofs < LSTM_BLOB_ELEMS * 2; ofs += 128 * 16) {
      async_copy_b128(lbase + ofs + tid * 16, gbase + ofs + tid * 16);
    }
  }
  for (int i = tid; i < 256; i += 128) sB[i] = bih[i] + bhh[i];
  wait_async0();
  __syncthreads();

  const int wave = tid >> 5;
  const int wg   = blockIdx.x * 4 + wave;
  if (wg >= ntiles) return;

  const int lane = tid & 31;
  const int half = lane >> 4;
  const int m    = lane & 15;
  const int nb   = wg * 16;

  const float* rx = x + (size_t)(nb + m) * DD;
  const float* ra = a + (size_t)(nb + m) * DD;

  // Pack A = [x | a] for this lane's row (4 K-chunks of 32)
  v16bf afr[4];
#pragma unroll
  for (int kc = 0; kc < 4; ++kc) {
    const float* base = (kc < 2) ? rx : ra;
    const int kb = (kc < 2) ? kc * 32 : (kc - 2) * 32;
    const float4 q0 = *(const float4*)(base + kb + half * 16);
    const float4 q1 = *(const float4*)(base + kb + half * 16 + 4);
    const float4 r0 = *(const float4*)(base + kb + half * 16 + 8);
    const float4 r1 = *(const float4*)(base + kb + half * 16 + 12);
    v16bf t;
    t[0]  = to_bf16(q0.x); t[1]  = to_bf16(q0.y); t[2]  = to_bf16(q0.z); t[3]  = to_bf16(q0.w);
    t[4]  = to_bf16(q1.x); t[5]  = to_bf16(q1.y); t[6]  = to_bf16(q1.z); t[7]  = to_bf16(q1.w);
    t[8]  = to_bf16(r0.x); t[9]  = to_bf16(r0.y); t[10] = to_bf16(r0.z); t[11] = to_bf16(r0.w);
    t[12] = to_bf16(r1.x); t[13] = to_bf16(r1.y); t[14] = to_bf16(r1.z); t[15] = to_bf16(r1.w);
    afr[kc] = t;
  }

  v8f zf;
#pragma unroll
  for (int j = 0; j < 8; ++j) zf[j] = 0.0f;

#pragma unroll
  for (int t = 0; t < 4; ++t) {
    v8f ai = zf, af = zf, ag = zf, ao = zf;
#pragma unroll
    for (int kc = 0; kc < 4; ++kc) {
      const int ko = kc * 32 + half * 16;
      const v16bf bi = *(const v16bf*)&sWt[((t)      * 16 + m) * 128 + ko];
      const v16bf bf = *(const v16bf*)&sWt[((4 + t)  * 16 + m) * 128 + ko];
      const v16bf bg = *(const v16bf*)&sWt[((8 + t)  * 16 + m) * 128 + ko];
      const v16bf bo = *(const v16bf*)&sWt[((12 + t) * 16 + m) * 128 + ko];
      ai = wmma_bf16(afr[kc], bi, ai);
      af = wmma_bf16(afr[kc], bf, af);
      ag = wmma_bf16(afr[kc], bg, ag);
      ao = wmma_bf16(afr[kc], bo, ao);
    }
    const float bi_ = sB[t * 16 + m];
    const float bf_ = sB[64 + t * 16 + m];
    const float bg_ = sB[128 + t * 16 + m];
    const float bo_ = sB[192 + t * 16 + m];
#pragma unroll
    for (int j = 0; j < 8; ++j) {
      const int row = j + 8 * half;
      const size_t off = (size_t)(nb + row) * DD + t * 16 + m;
      const float iv = fast_sigmoid(ai[j] + bi_);
      const float fv = fast_sigmoid(af[j] + bf_);
      const float gv = fast_tanh(ag[j] + bg_);
      const float ov = fast_sigmoid(ao[j] + bo_);
      const float cn = fv * c[off] + iv * gv;
      c[off] = cn;
      x[off] = ov * fast_tanh(cn);
    }
  }
}

// ---------------------------------------------------------------------------
// Readout: out[g] = sum_n sigmoid(x@gm_W + gm_b)[n,g] * (x@fm_W + fm_b)[n,g]
// Weight addresses are wave-uniform -> scalar (SMEM) loads.
// ---------------------------------------------------------------------------
__launch_bounds__(256)
__global__ void readout_kernel(const float* __restrict__ x,
                               const float* __restrict__ gmW,
                               const float* __restrict__ gmb,
                               const float* __restrict__ fmW,
                               const float* __restrict__ fmb,
                               float* __restrict__ out) {
  __shared__ float sacc[GG];
  if (threadIdx.x < GG) sacc[threadIdx.x] = 0.0f;
  __syncthreads();

  const int node = blockIdx.x * blockDim.x + threadIdx.x;
  if (node < NNODES) {
    const float* row = x + (size_t)node * DD;
    float xr[DD];
#pragma unroll
    for (int k = 0; k < DD; ++k) xr[k] = row[k];
    for (int g = 0; g < GG; ++g) {
      float dg = gmb[g], df = fmb[g];
#pragma unroll
      for (int k = 0; k < DD; ++k) {
        dg = __builtin_fmaf(xr[k], gmW[k * GG + g], dg);
        df = __builtin_fmaf(xr[k], fmW[k * GG + g], df);
      }
      atomicAdd(&sacc[g], fast_sigmoid(dg) * df);
    }
  }
  __syncthreads();
  if (threadIdx.x < GG) atomicAdd(&out[threadIdx.x], sacc[threadIdx.x]);
}

// ---------------------------------------------------------------------------
// Launcher
// ---------------------------------------------------------------------------
extern "C" void kernel_launch(void* const* d_in, const int* in_sizes, int n_in,
                              void* d_out, int out_size, void* d_ws, size_t ws_size,
                              hipStream_t stream) {
  const float* x_in      = (const float*)d_in[0];
  const float* edge_attr = (const float*)d_in[1];
  const int*   eidx      = (const int*)d_in[2];
  const int*   srcI      = eidx;
  const int*   dstI      = eidx + NEDGES;

  const float* feW1[2] = {(const float*)d_in[3],  (const float*)d_in[11]};
  const float* feB1[2] = {(const float*)d_in[4],  (const float*)d_in[12]};
  const float* feW2[2] = {(const float*)d_in[5],  (const float*)d_in[13]};
  const float* feB2[2] = {(const float*)d_in[6],  (const float*)d_in[14]};
  const float* Wih[2]  = {(const float*)d_in[7],  (const float*)d_in[15]};
  const float* Whh[2]  = {(const float*)d_in[8],  (const float*)d_in[16]};
  const float* bih[2]  = {(const float*)d_in[9],  (const float*)d_in[17]};
  const float* bhh[2]  = {(const float*)d_in[10], (const float*)d_in[18]};
  const float* gmW = (const float*)d_in[19];
  const float* gmb = (const float*)d_in[20];
  const float* fmW = (const float*)d_in[21];
  const float* fmb = (const float*)d_in[22];

  const int nstate = NNODES * DD;               // 3.2M floats
  float* xcur = (float*)d_ws;
  float* cbuf = xcur + (size_t)nstate;
  float* agg  = cbuf + (size_t)nstate;
  __bf16* wb  = (__bf16*)(agg + (size_t)nstate);
  __bf16* feblob[2]   = {wb, wb + FE_BLOB_ELEMS};
  __bf16* lstmblob[2] = {wb + 2 * FE_BLOB_ELEMS,
                         wb + 2 * FE_BLOB_ELEMS + LSTM_BLOB_ELEMS};

  const int elem_blocks = (nstate + 255) / 256; // 12500

  init_state_kernel<<<elem_blocks, 256, 0, stream>>>(x_in, xcur, cbuf, nstate);
  for (int s = 0; s < 2; ++s) {
    convert_weights_kernel<<<(LSTM_BLOB_ELEMS + 255) / 256, 256, 0, stream>>>(
        feW1[s], feW2[s], Wih[s], Whh[s], feblob[s], lstmblob[s]);
  }

  const int edge_tiles = NEDGES / 16;           // 50000 waves, exact
  const int edge_blocks = edge_tiles / 4;       // 12500 blocks of 4 waves
  const int node_tiles = NNODES / 16;           // 3125 waves, exact
  const int lstm_blocks = (node_tiles + 3) / 4; // 782

  for (int s = 0; s < 2; ++s) {
    zero_kernel<<<elem_blocks, 256, 0, stream>>>(agg, nstate);
    edge_mpn_kernel<<<edge_blocks, 128, 0, stream>>>(
        xcur, edge_attr, srcI, dstI, feblob[s], feB1[s], feB2[s], agg);
    lstm_kernel<<<lstm_blocks, 128, 0, stream>>>(
        xcur, agg, cbuf, lstmblob[s], bih[s], bhh[s], node_tiles);
  }

  zero_kernel<<<1, 64, 0, stream>>>((float*)d_out, out_size);
  readout_kernel<<<(NNODES + 255) / 256, 256, 0, stream>>>(
      xcur, gmW, gmb, fmW, fmb, (float*)d_out);
}
